// DIFLayer_63273458204748
// MI455X (gfx1250) — matrix operand order repaired and checked by reading, not compiled, not run on hardware
//
#include <hip/hip_runtime.h>
#include <hip/hip_bf16.h>
#include <math.h>

typedef __attribute__((ext_vector_type(16))) _Float16 v16h;
typedef __attribute__((ext_vector_type(8)))  float    v8f;

#define PP      64
#define KCOMP   64
#define H1C     256
#define H2C     256
#define LOG2PI  1.8378770664093453f

// LDS strides (in f16 elements) — padded for bank-conflict-free b128 fragment loads
#define W1S 72     // rows of 64 -> 72  (36 dwords, 36h%64 distinct for h=0..15)
#define W2S 264    // rows of 256 -> 264 (132 dwords, 4h%64 distinct)
#define W3S 264
#define HBS 264    // per-wave hidden buffer stride
#define ZS  72     // per-wave z buffer stride
#define LGS 68     // per-wave logits stride (f32)

// LDS carve-up (bytes), all sections 16B aligned
#define OFF_W1  0
#define SZ_W1   (H1C * W1S * 2)          // 36864
#define OFF_W2  (OFF_W1 + SZ_W1)
#define SZ_W2   (H2C * W2S * 2)          // 135168
#define OFF_W3  (OFF_W2 + SZ_W2)
#define SZ_W3   (KCOMP * W3S * 2)        // 33792
#define OFF_HB  (OFF_W3 + SZ_W3)
#define SZ_HB_W (16 * HBS * 2)           // 8448 per wave
#define OFF_SB  (OFF_HB + 8 * SZ_HB_W)
#define SZ_SB_W (16 * LGS * 4)           // 4352 per wave (z f16 16x72 = 2304 fits too)
#define OFF_RED (OFF_SB + 8 * SZ_SB_W)
#define SZ_RED  (8 * 16 * 2 * 4)         // 1024
#define OFF_B1  (OFF_RED + SZ_RED)       // biases staged as f32
#define OFF_B2  (OFF_B1 + H1C * 4)
#define OFF_B3  (OFF_B2 + H2C * 4)
#define LDS_TOTAL (OFF_B3 + KCOMP * 4)   // 311552 <= 320KB

// Hardware tanh (CDNA5 TRANS op) — 1 instruction vs ~25 for OCML tanhf
__device__ __forceinline__ float fast_tanh(float v) {
#if __has_builtin(__builtin_amdgcn_tanhf)
    return __builtin_amdgcn_tanhf(v);
#else
    float r;
    asm("v_tanh_f32 %0, %1" : "=v"(r) : "v"(v));
    return r;
#endif
}

// ---- WMMA fragment loaders (CDNA5 wave32 16x16x32 f16 layouts) ----
// A (16x32, 16-bit): lane L holds row M=L&15; VGPR0..3 -> K = hi..hi+7,
// VGPR4..7 -> K = 16+hi..16+hi+7, hi = (L>=16)?8:0.  => two contiguous b128s.
__device__ __forceinline__ v16h load_frag_A(const _Float16* rowbase, int kbase, int lane) {
    const int hi = (lane >> 4) * 8;
    union { int4 q[2]; v16h h; } u;
    u.q[0] = *(const int4*)(rowbase + kbase + hi);
    u.q[1] = *(const int4*)(rowbase + kbase + hi + 16);
    return u.h;
}
// B (32x16, 16-bit): lane L holds column N=L&15; VGPR0..3 -> K = hi..hi+7,
// VGPR4..7 -> K = 16+hi..16+hi+7, hi = (L>=16)?8:0.  => two contiguous b128s.
__device__ __forceinline__ v16h load_frag_B(const _Float16* colbase, int kbase, int lane) {
    const int off = (lane >> 4) * 8;
    union { int4 q[2]; v16h h; } u;
    u.q[0] = *(const int4*)(colbase + kbase + off);
    u.q[1] = *(const int4*)(colbase + kbase + off + 16);
    return u.h;
}

__global__ __launch_bounds__(256, 1)
void dif_fused_wmma(const float* __restrict__ x,  const float* __restrict__ mloc,
                    const float* __restrict__ lgs,
                    const float* __restrict__ W1, const float* __restrict__ b1,
                    const float* __restrict__ W2, const float* __restrict__ b2,
                    const float* __restrict__ W3, const float* __restrict__ b3,
                    float* __restrict__ out)
{
    extern __shared__ char lds_raw[];
    _Float16* w1 = (_Float16*)(lds_raw + OFF_W1);
    _Float16* w2 = (_Float16*)(lds_raw + OFF_W2);
    _Float16* w3 = (_Float16*)(lds_raw + OFF_W3);
    const int tid  = threadIdx.x;
    const int wave = tid >> 5;
    const int lane = tid & 31;
    _Float16* hb  = (_Float16*)(lds_raw + OFF_HB + wave * SZ_HB_W);
    _Float16* zb  = (_Float16*)(lds_raw + OFF_SB + wave * SZ_SB_W);
    float*    lg  = (float*)   (lds_raw + OFF_SB + wave * SZ_SB_W);  // reuses z scratch
    float*    red = (float*)   (lds_raw + OFF_RED);
    float*    bb1 = (float*)   (lds_raw + OFF_B1);
    float*    bb2 = (float*)   (lds_raw + OFF_B2);
    float*    bb3 = (float*)   (lds_raw + OFF_B3);

    // Stage all MLP weights into LDS as f16 (whole MLP fits: ~206KB) + biases as f32
    for (int i = tid; i < H1C * PP;   i += 256) w1[(i >> 6) * W1S + (i & 63)]  = (_Float16)W1[i];
    for (int i = tid; i < H2C * H1C;  i += 256) w2[(i >> 8) * W2S + (i & 255)] = (_Float16)W2[i];
    for (int i = tid; i < KCOMP * H2C; i += 256) w3[(i >> 8) * W3S + (i & 255)] = (_Float16)W3[i];
    if (tid < H1C)  bb1[tid] = b1[tid];
    if (tid < H2C)  bb2[tid] = b2[tid];
    if (tid < KCOMP) bb3[tid] = b3[tid];
    __syncthreads();

    const int n0   = blockIdx.x * 16;
    const int mrow = lane & 15;   // row within tile / column within output tile
    const int half = lane >> 4;

    const v8f vzero = {0.f,0.f,0.f,0.f,0.f,0.f,0.f,0.f};
    float rmax = -INFINITY, rsum = 0.f;   // streaming logsumexp over k (lanes 0..15)

    for (int iter = 0; iter < 8; ++iter) {
        const int k = wave + 8 * iter;

        // ---- z tile: z[m,p] = (x[n0+m,p] - mu[k,p]) * exp(-log_s[k,p]); also sum z^2, sum log_s
        const float* xr = x    + (size_t)(n0 + mrow) * PP + half * 32;
        const float* mr = mloc + (size_t)k * PP + half * 32;
        const float* lr = lgs  + (size_t)k * PP + half * 32;
        _Float16*    zr = zb + mrow * ZS + half * 32;
        float ssq = 0.f, sls = 0.f;
        #pragma unroll
        for (int p = 0; p < 32; ++p) {
            const float ls = lr[p];
            const float zv = (xr[p] - mr[p]) * __expf(-ls);
            ssq += zv * zv; sls += ls;
            zr[p] = (_Float16)zv;
        }
        ssq += __shfl_xor(ssq, 16, 32);
        sls += __shfl_xor(sls, 16, 32);
        const float ref_lp  = -0.5f * ssq - 0.5f * (float)PP * LOG2PI;
        const float log_det = -sls;

        // ---- layer 1: [16x64] @ [64x256] -> acc[16] tiles
        v8f acc[16];
        #pragma unroll
        for (int t = 0; t < 16; ++t) acc[t] = vzero;
        const _Float16* zrowA = zb + mrow * ZS;
        #pragma unroll
        for (int kk = 0; kk < 2; ++kk) {
            const v16h A = load_frag_A(zrowA, kk * 32, lane);
            #pragma unroll
            for (int t = 0; t < 16; ++t) {
                const v16h B = load_frag_B(w1 + (t * 16 + mrow) * W1S, kk * 32, lane);
                acc[t] = __builtin_amdgcn_wmma_f32_16x16x32_f16(
                    false, A, false, B, (short)0, acc[t], false, false);
            }
        }
        #pragma unroll
        for (int t = 0; t < 16; ++t) {
            const int   h  = t * 16 + mrow;
            const float bv = bb1[h];
            #pragma unroll
            for (int r = 0; r < 8; ++r) {
                const int M = r + half * 8;
                hb[M * HBS + h] = (_Float16)fast_tanh(acc[t][r] + bv);
            }
        }

        // ---- layer 2: [16x256] @ [256x256] -> acc[16] (acc in regs, hb overwritten after)
        #pragma unroll
        for (int t = 0; t < 16; ++t) acc[t] = vzero;
        const _Float16* hrowA = hb + mrow * HBS;
        #pragma unroll
        for (int kk = 0; kk < 8; ++kk) {
            const v16h A = load_frag_A(hrowA, kk * 32, lane);
            #pragma unroll
            for (int t = 0; t < 16; ++t) {
                const v16h B = load_frag_B(w2 + (t * 16 + mrow) * W2S, kk * 32, lane);
                acc[t] = __builtin_amdgcn_wmma_f32_16x16x32_f16(
                    false, A, false, B, (short)0, acc[t], false, false);
            }
        }
        #pragma unroll
        for (int t = 0; t < 16; ++t) {
            const int   h  = t * 16 + mrow;
            const float bv = bb2[h];
            #pragma unroll
            for (int r = 0; r < 8; ++r) {
                const int M = r + half * 8;
                hb[M * HBS + h] = (_Float16)fast_tanh(acc[t][r] + bv);
            }
        }

        // ---- layer 3: [16x256] @ [256x64] -> 4 tiles of logits
        v8f acc3[4];
        #pragma unroll
        for (int t = 0; t < 4; ++t) acc3[t] = vzero;
        #pragma unroll
        for (int kk = 0; kk < 8; ++kk) {
            const v16h A = load_frag_A(hrowA, kk * 32, lane);
            #pragma unroll
            for (int t = 0; t < 4; ++t) {
                const v16h B = load_frag_B(w3 + (t * 16 + mrow) * W3S, kk * 32, lane);
                acc3[t] = __builtin_amdgcn_wmma_f32_16x16x32_f16(
                    false, A, false, B, (short)0, acc3[t], false, false);
            }
        }
        #pragma unroll
        for (int t = 0; t < 4; ++t) {
            const int   c  = t * 16 + mrow;
            const float bv = bb3[c];
            #pragma unroll
            for (int r = 0; r < 8; ++r) {
                const int M = r + half * 8;
                lg[M * LGS + c] = acc3[t][r] + bv;   // z buffer is dead; reuse as f32 logits
            }
        }

        // ---- diag log-softmax over the 64 logits of row `mrow`, split across lane halves
        {
            const float* row = lg + mrow * LGS + half * 32;
            float mx = -INFINITY;
            #pragma unroll
            for (int c = 0; c < 32; ++c) mx = fmaxf(mx, row[c]);
            mx = fmaxf(mx, __shfl_xor(mx, 16, 32));
            float s = 0.f;
            #pragma unroll
            for (int c = 0; c < 32; ++c) s += __expf(row[c] - mx);
            s += __shfl_xor(s, 16, 32);
            const float lse  = mx + __logf(s);
            const float diag = lg[mrow * LGS + k];
            const float lv   = ref_lp + (diag - lse) + log_det;
            if (lane < 16) {   // streaming logsumexp over k, one row per lane
                const float nm = fmaxf(rmax, lv);
                rsum = rsum * __expf(rmax - nm) + __expf(lv - nm);
                rmax = nm;
            }
        }
    }

    // ---- cross-wave merge of (rmax, rsum) pairs, then final out[n] = logsumexp_k
    if (lane < 16) {
        red[(wave * 16 + lane) * 2 + 0] = rmax;
        red[(wave * 16 + lane) * 2 + 1] = rsum;
    }
    __syncthreads();
    if (tid < 16) {
        float Mx = -INFINITY, S = 0.f;
        #pragma unroll
        for (int w = 0; w < 8; ++w) {
            const float a = red[(w * 16 + tid) * 2 + 0];
            const float b = red[(w * 16 + tid) * 2 + 1];
            const float nm = fmaxf(Mx, a);
            S = S * __expf(Mx - nm) + b * __expf(a - nm);
            Mx = nm;
        }
        out[n0 + tid] = Mx + __logf(S);
    }
}

extern "C" void kernel_launch(void* const* d_in, const int* in_sizes, int n_in,
                              void* d_out, int out_size, void* d_ws, size_t ws_size,
                              hipStream_t stream) {
    const float* x    = (const float*)d_in[0];
    const float* m    = (const float*)d_in[1];
    const float* lgsc = (const float*)d_in[2];
    const float* W1   = (const float*)d_in[3];
    const float* b1   = (const float*)d_in[4];
    const float* W2   = (const float*)d_in[5];
    const float* b2   = (const float*)d_in[6];
    const float* W3   = (const float*)d_in[7];
    const float* b3   = (const float*)d_in[8];
    float* out = (float*)d_out;

    const int N = in_sizes[0] / PP;     // 16384
    const int blocks = N / 16;          // 1024 row-tiles

    dif_fused_wmma<<<blocks, 256, LDS_TOTAL, stream>>>(
        x, m, lgsc, W1, b1, W2, b2, W3, b3, out);
}
